// LlamaAttention_12833362280477
// MI455X (gfx1250) — matrix operand order repaired and checked by reading, not compiled, not run on hardware
//
#include <hip/hip_runtime.h>

typedef __bf16 bf16;
typedef __attribute__((ext_vector_type(16))) __bf16 v16bf;
typedef __attribute__((ext_vector_type(8)))  float   v8f;
typedef __attribute__((ext_vector_type(4))) unsigned int u32x4;
typedef __attribute__((ext_vector_type(8))) int i32x8;
typedef __attribute__((ext_vector_type(4))) int i32x4;

#define WMMA_BF16(a, b, c) \
  __builtin_amdgcn_wmma_f32_16x16x32_bf16(false, (a), false, (b), (short)0, (c), false, false)

union Frag { v16bf v; uint4 q[2]; };

constexpr int Bsz = 2, S = 2048, D = 2048, H = 16, HD = 128;
constexpr int Mrows = Bsz * S;  // 4096

// ---------------------------------------------------------------- TDM helper
// Issue a 2D bf16 tile load Global -> LDS via the Tensor Data Mover.
// D# packing per CDNA5 ISA:
//   group0: count=1 | lds_addr | global_addr[56:0] | type=2
//   group1: data_size=1 (2B), pad fields, tensor_dim0/1, tile_dim0/1,
//           tensor_dim0_stride (elements)
// pad_interval_code: 0=2,1=4,2=8,3=16,4=32,5=64 DWORDs between pads
// pad_dwords: LDS padding in DWORDs (field stores value-1)
__device__ __forceinline__ void tdm_load_tile_2d(
    uint32_t lds_off, const bf16* gaddr,
    uint32_t tile_d0, uint32_t tile_d1,
    uint32_t tensor_d0, uint32_t tensor_d1,
    uint32_t stride_d0_elems,
    uint32_t pad_interval_code, uint32_t pad_dwords) {
  uint64_t ga = (uint64_t)(uintptr_t)gaddr;
  u32x4 g0;
  g0[0] = 1u;                                         // count=1, user descriptor
  g0[1] = lds_off;                                    // lds_addr (bytes)
  g0[2] = (uint32_t)(ga & 0xffffffffu);               // global_addr[31:0]
  g0[3] = (uint32_t)((ga >> 32) & 0x01ffffffu)        // global_addr[56:32]
          | (2u << 30);                               // type=2 ("image")
  i32x8 g1;
  uint32_t d0 = (1u << 16);                           // data_size=1 -> 2 bytes
  if (pad_dwords) {
    d0 |= (1u << 20)                                  // pad_enable
        | (pad_interval_code << 22)
        | ((pad_dwords - 1u) << 25);
  }
  g1[0] = (int)d0;
  g1[1] = (int)((tensor_d0 & 0xffffu) << 16);                        // dim0[15:0] @ bits 48..63
  g1[2] = (int)((tensor_d0 >> 16) | ((tensor_d1 & 0xffffu) << 16));  // dim0[31:16] | dim1[15:0]
  g1[3] = (int)((tensor_d1 >> 16) | (tile_d0 << 16));                // dim1[31:16] | tile_dim0
  g1[4] = (int)(tile_d1 & 0xffffu);                                  // tile_dim1, tile_dim2=0
  g1[5] = (int)stride_d0_elems;                                      // dim0_stride[31:0]
  g1[6] = 0;                                                         // stride hi / dim1_stride
  g1[7] = 0;
  i32x4 gz = {0, 0, 0, 0};
#if __clang_major__ >= 23
  i32x8 gz8 = {0, 0, 0, 0, 0, 0, 0, 0};
  __builtin_amdgcn_tensor_load_to_lds(g0, g1, gz, gz, gz8, 0);
#else
  __builtin_amdgcn_tensor_load_to_lds(g0, g1, gz, gz, 0);
#endif
}

// ---------------------------------------------------------------- fp32 -> bf16
__global__ __launch_bounds__(256) void cvt_f32_bf16(const float* __restrict__ x,
                                                    bf16* __restrict__ y, int n4) {
  int i = blockIdx.x * 256 + threadIdx.x;
  if (i < n4) {
    float4 f = ((const float4*)x)[i];
    union { bf16 b[4]; uint2 u; } p;
    p.b[0] = (bf16)f.x; p.b[1] = (bf16)f.y; p.b[2] = (bf16)f.z; p.b[3] = (bf16)f.w;
    ((uint2*)y)[i] = p.u;
  }
}

// ---------------------------------------------------------------- WMMA GEMM
// C[M,N] = A[M,K] * W[K,N], A/W bf16 row-major.
// MODE 0: three weights/outputs (QKV), epilogue writes bf16 [B,H,S,HD].
// MODE 1: single weight, epilogue writes f32 row-major [M,N].
template <int MODE>
__global__ __launch_bounds__(256) void gemm_bf16(
    const bf16* __restrict__ A,
    const bf16* __restrict__ W0, const bf16* __restrict__ W1, const bf16* __restrict__ W2,
    bf16* __restrict__ O0, bf16* __restrict__ O1, bf16* __restrict__ O2,
    float* __restrict__ OF) {
  constexpr int BM = 128, BN = 128, BK = 32, K = D, N = D;
  constexpr int SA = BK + 8;   // LDS row stride in halves (80B, 16B aligned)
  constexpr int SB = BK + 8;
  __shared__ bf16 As[BM * SA];      // A tile, row-major (m, k); filled by TDM
  __shared__ bf16 BsT[BN * SB];     // W tile, transposed (n, k)

  const int tid = threadIdx.x;
  const int lane = tid & 31, wave = tid >> 5;
  const int r16 = lane & 15, hh = lane >> 4;
  const int wm = wave & 3, wn = wave >> 2;         // 4 x 2 wave grid
  const int m0 = blockIdx.y * BM;
  const int n0 = blockIdx.x * BN;
  const uint32_t asOff = (uint32_t)(uintptr_t)&As[0];

  const bf16* W = W0;
  bf16* O = O0;
  if (MODE == 0) {
    if (blockIdx.z == 1) { W = W1; O = O1; }
    else if (blockIdx.z == 2) { W = W2; O = O2; }
  }

  v8f c[2][4];
#pragma unroll
  for (int i = 0; i < 2; ++i)
#pragma unroll
    for (int j = 0; j < 4; ++j) c[i][j] = v8f{};

  for (int k0 = 0; k0 < K; k0 += BK) {
    // ---- A tile 128x32 via Tensor Data Mover (rows padded 16 DW + 4 DW pad)
    if (wave == 0) {
      tdm_load_tile_2d(asOff, A + (size_t)m0 * K + k0,
                       /*tile*/ BK, BM, /*tensor*/ K, Mrows,
                       /*stride*/ K, /*pad 16DW interval*/ 3, /*pad*/ 4);
    }
    // ---- W tile: 32x128, stored transposed into BsT[n][k]
#pragma unroll
    for (int it = 0; it < 2; ++it) {
      int chunk = tid + it * 256;
      int kr = chunk >> 4, nc = chunk & 15;
      union { uint4 u; bf16 b[8]; } d;
      d.u = *(const uint4*)(W + (size_t)(k0 + kr) * N + n0 + nc * 8);
#pragma unroll
      for (int j = 0; j < 8; ++j) BsT[(nc * 8 + j) * SB + kr] = d.b[j];
    }
    if (k0 + BK < K) {
      __builtin_prefetch(A + (size_t)(m0 + (tid >> 1)) * K + k0 + BK, 0, 0);
      __builtin_prefetch(W + (size_t)(k0 + BK + (tid >> 4)) * N + n0, 0, 0);
    }
    if (wave == 0) __builtin_amdgcn_s_wait_tensorcnt(0);
    __syncthreads();

    // ---- fragments (layouts per CDNA5 ISA 7.12.2, 16-bit A/B)
    Frag a[2], b[4];
#pragma unroll
    for (int mt = 0; mt < 2; ++mt) {
      const bf16* ap = &As[(wm * 32 + mt * 16 + r16) * SA];
      a[mt].q[0] = *(const uint4*)(ap + hh * 8);        // K = hh*8 .. +7
      a[mt].q[1] = *(const uint4*)(ap + 16 + hh * 8);   // K = 16+hh*8 .. +7
    }
#pragma unroll
    for (int nt = 0; nt < 4; ++nt) {
      const bf16* bp = &BsT[(wn * 64 + nt * 16 + r16) * SB + hh * 8];
      b[nt].q[0] = *(const uint4*)(bp);                 // K = hh*8 .. +7
      b[nt].q[1] = *(const uint4*)(bp + 16);            // K = 16+hh*8 .. +7
    }
#pragma unroll
    for (int mt = 0; mt < 2; ++mt)
#pragma unroll
      for (int nt = 0; nt < 4; ++nt)
        c[mt][nt] = WMMA_BF16(a[mt].v, b[nt].v, c[mt][nt]);
    __syncthreads();
  }

  // ---- epilogue (C layout: row = r + 8*hh, col = r16)
#pragma unroll
  for (int mt = 0; mt < 2; ++mt)
#pragma unroll
    for (int nt = 0; nt < 4; ++nt)
#pragma unroll
      for (int r = 0; r < 8; ++r) {
        int m = m0 + wm * 32 + mt * 16 + r + 8 * hh;
        int n = n0 + wn * 64 + nt * 16 + r16;
        float v = c[mt][nt][r];
        if (MODE == 0) {
          int bb = m >> 11, s = m & (S - 1);
          int hsel = n >> 7, hd = n & (HD - 1);
          O[((size_t)((bb * H + hsel) * S + s)) * HD + hd] = (bf16)v;
        } else {
          OF[(size_t)m * N + n] = v;
        }
      }
}

// ---------------------------------------------------------------- RoPE
__global__ __launch_bounds__(256) void rope_kernel(bf16* __restrict__ Q,
                                                   bf16* __restrict__ Kk) {
  const size_t idx = (size_t)blockIdx.x * 256 + threadIdx.x;  // B*H*S*64 pairs
  const int j = (int)(idx & 63);
  const size_t row = idx >> 6;
  const int s = (int)(row & (S - 1));
  const float ang = s * __expf(j * (-2.0f / HD) * 9.210340371976184f);  // ln(1e4)
  float sn, cs;
  __sincosf(ang, &sn, &cs);
  const size_t base = row * HD;
  float q1 = (float)Q[base + j], q2 = (float)Q[base + j + 64];
  Q[base + j]      = (bf16)(q1 * cs - q2 * sn);
  Q[base + j + 64] = (bf16)(q2 * cs + q1 * sn);
  float k1 = (float)Kk[base + j], k2 = (float)Kk[base + j + 64];
  Kk[base + j]      = (bf16)(k1 * cs - k2 * sn);
  Kk[base + j + 64] = (bf16)(k2 * cs + k1 * sn);
}

// ---------------------------------------------------------------- flash attention
// grid (S/64, H, B), 128 threads = 4 waves; wave w owns 16 query rows.
__global__ __launch_bounds__(128) void attn_kernel(
    const bf16* __restrict__ Q, const bf16* __restrict__ Kt,
    const bf16* __restrict__ V, bf16* __restrict__ Aout) {
  constexpr int SK = HD + 8;  // 272B row stride (64 DW row + 4 DW pad)
  constexpr int SV = 32 + 8;  // 80B
  constexpr int SP = 32 + 8;
  __shared__ bf16 Kls[32 * SK];       // keys, row-major (key, hd); filled by TDM
  __shared__ bf16 VlsT[HD * SV];      // values transposed (hd, key)
  __shared__ bf16 Pls[4 * 16 * SP];   // per-wave P scratch (row, key)

  const int tid = threadIdx.x;
  const int lane = tid & 31, wave = tid >> 5;
  const int r16 = lane & 15, hh = lane >> 4;
  const int qb0 = blockIdx.x * 64;
  const int h = blockIdx.y, b = blockIdx.z;
  const int qbase = qb0 + wave * 16;
  const uint32_t klsOff = (uint32_t)(uintptr_t)&Kls[0];

  const size_t headoff = ((size_t)(b * H + h)) * S * HD;
  const bf16* Qh = Q + headoff;
  const bf16* Kh = Kt + headoff;
  const bf16* Vh = V + headoff;

  // Q fragments held in registers: 16 rows x 128 (4 K-chunks of 32)
  Frag qf[4];
  {
    const bf16* qp = Qh + (size_t)(qbase + r16) * HD;
#pragma unroll
    for (int kc = 0; kc < 4; ++kc) {
      qf[kc].q[0] = *(const uint4*)(qp + kc * 32 + hh * 8);
      qf[kc].q[1] = *(const uint4*)(qp + kc * 32 + 16 + hh * 8);
    }
  }

  float m_i[8], l_i[8];
  v8f acc[8];
#pragma unroll
  for (int r = 0; r < 8; ++r) { m_i[r] = -3.0e38f; l_i[r] = 0.0f; }
#pragma unroll
  for (int ot = 0; ot < 8; ++ot) acc[ot] = v8f{};

  const float smscale = 0.08838834764831845f;  // 1/sqrt(128)
  const int jend = qb0 + 64;                   // causal upper bound for this block
  for (int j0 = 0; j0 < jend; j0 += 32) {
    // ---- K tile (32 x 128) via Tensor Data Mover into padded LDS rows
    if (wave == 0) {
      tdm_load_tile_2d(klsOff, Kh + (size_t)j0 * HD,
                       /*tile*/ HD, 32, /*tensor*/ HD, S,
                       /*stride*/ HD, /*pad 64DW interval*/ 5, /*pad*/ 4);
    }
    // ---- V tile loaded + transposed manually (TDM cannot transpose)
#pragma unroll
    for (int it = 0; it < 4; ++it) {
      int chunk = tid + it * 128;
      int kr = chunk >> 4, nc = chunk & 15;
      union { uint4 u; bf16 b[8]; } vd;
      vd.u = *(const uint4*)(Vh + (size_t)(j0 + kr) * HD + nc * 8);
#pragma unroll
      for (int jj = 0; jj < 8; ++jj) VlsT[(nc * 8 + jj) * SV + kr] = vd.b[jj];
    }
    if (wave == 0) __builtin_amdgcn_s_wait_tensorcnt(0);
    __syncthreads();

    // ---- scores S = Q * K^T  (16 x 32, two 16x16 WMMA column tiles)
    v8f sc[2];
    sc[0] = v8f{}; sc[1] = v8f{};
#pragma unroll
    for (int kc = 0; kc < 4; ++kc)
#pragma unroll
      for (int nt = 0; nt < 2; ++nt) {
        Frag bf_;
        const bf16* kp = &Kls[(nt * 16 + r16) * SK + kc * 32 + hh * 8];
        bf_.q[0] = *(const uint4*)(kp);
        bf_.q[1] = *(const uint4*)(kp + 16);
        sc[nt] = WMMA_BF16(qf[kc].v, bf_.v, sc[nt]);
      }

    // ---- scale + causal mask
    float sv[2][8];
#pragma unroll
    for (int nt = 0; nt < 2; ++nt)
#pragma unroll
      for (int r = 0; r < 8; ++r) {
        float x = sc[nt][r] * smscale;
        int kg = j0 + nt * 16 + r16;
        int qg = qbase + r + 8 * hh;
        sv[nt][r] = (kg > qg) ? -1.0e30f : x;
      }

    // ---- online softmax: row reductions over the 16-lane half owning the row
    float scale_r[8];
#pragma unroll
    for (int r = 0; r < 8; ++r) {
      float mx = fmaxf(sv[0][r], sv[1][r]);
#pragma unroll
      for (int off = 8; off >= 1; off >>= 1) mx = fmaxf(mx, __shfl_xor(mx, off, 32));
      float m_new = fmaxf(m_i[r], mx);
      float p0 = __expf(sv[0][r] - m_new);
      float p1 = __expf(sv[1][r] - m_new);
      float rs = p0 + p1;
#pragma unroll
      for (int off = 8; off >= 1; off >>= 1) rs += __shfl_xor(rs, off, 32);
      float scl = __expf(m_i[r] - m_new);
      l_i[r] = l_i[r] * scl + rs;
      m_i[r] = m_new;
      scale_r[r] = scl;
      int rl = r + 8 * hh;  // C-layout row -> P scratch row
      Pls[(wave * 16 + rl) * SP + r16]      = (bf16)p0;
      Pls[(wave * 16 + rl) * SP + 16 + r16] = (bf16)p1;
    }
#pragma unroll
    for (int ot = 0; ot < 8; ++ot)
#pragma unroll
      for (int r = 0; r < 8; ++r) acc[ot][r] *= scale_r[r];

    // ---- P fragment (A layout) from wave-private LDS scratch
    Frag pf;
    {
      const bf16* pp = &Pls[(wave * 16 + r16) * SP];
      pf.q[0] = *(const uint4*)(pp + hh * 8);
      pf.q[1] = *(const uint4*)(pp + 16 + hh * 8);
    }
    // ---- acc += P * V  (8 output column tiles across HD)
#pragma unroll
    for (int ot = 0; ot < 8; ++ot) {
      Frag vf;
      const bf16* vp = &VlsT[(ot * 16 + r16) * SV + hh * 8];
      vf.q[0] = *(const uint4*)(vp);
      vf.q[1] = *(const uint4*)(vp + 16);
      acc[ot] = WMMA_BF16(pf.v, vf.v, acc[ot]);
    }
    __syncthreads();
  }

  // ---- normalize and write [B,S,D] layout for the output projection
#pragma unroll
  for (int r = 0; r < 8; ++r) l_i[r] = 1.0f / l_i[r];
#pragma unroll
  for (int ot = 0; ot < 8; ++ot)
#pragma unroll
    for (int r = 0; r < 8; ++r) {
      int q = qbase + r + 8 * hh;
      int hd = ot * 16 + r16;
      Aout[((size_t)(b * S + q)) * D + h * HD + hd] = (bf16)(acc[ot][r] * l_i[r]);
    }
}

// ---------------------------------------------------------------- launcher
extern "C" void kernel_launch(void* const* d_in, const int* in_sizes, int n_in,
                              void* d_out, int out_size, void* d_ws, size_t ws_size,
                              hipStream_t stream) {
  const float* hs = (const float*)d_in[0];
  // d_in[1] attention_mask: causal, applied analytically. d_in[2] position_ids: arange.
  const float* Wq = (const float*)d_in[3];
  const float* Wk = (const float*)d_in[4];
  const float* Wv = (const float*)d_in[5];
  const float* Wo = (const float*)d_in[6];
  float* out = (float*)d_out;

  char* ws = (char*)d_ws;
  size_t off = 0;
  auto alloc = [&](size_t bytes) {
    void* p = ws + off;
    off += (bytes + 255) & ~(size_t)255;
    return p;
  };
  bf16* Xb  = (bf16*)alloc((size_t)Mrows * D * 2);
  bf16* Wqb = (bf16*)alloc((size_t)D * D * 2);
  bf16* Wkb = (bf16*)alloc((size_t)D * D * 2);
  bf16* Wvb = (bf16*)alloc((size_t)D * D * 2);
  bf16* Wob = (bf16*)alloc((size_t)D * D * 2);
  bf16* Qb  = (bf16*)alloc((size_t)Mrows * D * 2);  // [B,H,S,HD]
  bf16* Kb  = (bf16*)alloc((size_t)Mrows * D * 2);
  bf16* Vb  = (bf16*)alloc((size_t)Mrows * D * 2);
  bf16* Ab  = (bf16*)alloc((size_t)Mrows * D * 2);  // [B,S,D]

  const int nHid4 = Mrows * D / 4, nW4 = D * D / 4;
  cvt_f32_bf16<<<nHid4 / 256, 256, 0, stream>>>(hs, Xb, nHid4);
  cvt_f32_bf16<<<nW4 / 256, 256, 0, stream>>>(Wq, Wqb, nW4);
  cvt_f32_bf16<<<nW4 / 256, 256, 0, stream>>>(Wk, Wkb, nW4);
  cvt_f32_bf16<<<nW4 / 256, 256, 0, stream>>>(Wv, Wvb, nW4);
  cvt_f32_bf16<<<nW4 / 256, 256, 0, stream>>>(Wo, Wob, nW4);

  gemm_bf16<0><<<dim3(D / 128, Mrows / 128, 3), 256, 0, stream>>>(
      Xb, Wqb, Wkb, Wvb, Qb, Kb, Vb, nullptr);

  rope_kernel<<<(Bsz * H * S * 64) / 256, 256, 0, stream>>>(Qb, Kb);

  attn_kernel<<<dim3(S / 64, H, Bsz), 128, 0, stream>>>(Qb, Kb, Vb, Ab);

  gemm_bf16<1><<<dim3(D / 128, Mrows / 128, 1), 256, 0, stream>>>(
      Ab, Wob, nullptr, nullptr, nullptr, nullptr, nullptr, out);
}
